// EncoderBlock_17961553231939
// MI455X (gfx1250) — matrix operand order repaired and checked by reading, not compiled, not run on hardware
//
#include <hip/hip_runtime.h>
#include <hip/hip_bf16.h>
#include <math.h>

typedef __attribute__((ext_vector_type(16))) _Float16 v16h;
typedef __attribute__((ext_vector_type(8)))  _Float16 v8h;
typedef __attribute__((ext_vector_type(4)))  _Float16 v4h;
typedef __attribute__((ext_vector_type(8)))  float    v8f;
typedef __attribute__((ext_vector_type(4)))  unsigned int v4u;
typedef __attribute__((ext_vector_type(8)))  int      v8i;
typedef __attribute__((ext_vector_type(4)))  int      v4i;

#define BDIM   8
#define SQ     1024
#define HDIM   512
#define HEADS  16
#define DEPTH  32
#define ROWS   (BDIM*SQ)        // 8192
#define IN_EPS     1e-05f
#define DEMOD_EPS  1e-08f
#define CLAMPV     256.0f
#define LRELU      0.2f

// ---------------------------------------------------------------------------
// WMMA fragment loaders (CDNA5 16x16x32 f16 layouts)
// A: lane -> M row (lane%16); elements e<8 -> K = half*8+e ; e>=8 -> K = 16+half*8+(e-8)
// B: lane -> N col (lane%16); elements e<8 -> K = half*8+e ; e>=8 -> K = 16+half*8+(e-8)
// ---------------------------------------------------------------------------
__device__ __forceinline__ v16h ldA(const _Float16* rowbase, int half) {
    const v8h lo = *(const v8h*)(rowbase + half * 8);
    const v8h hi = *(const v8h*)(rowbase + 16 + half * 8);
    v16h r;
#pragma unroll
    for (int i = 0; i < 8; ++i) { r[i] = lo[i]; r[i + 8] = hi[i]; }
    return r;
}

__device__ __forceinline__ v16h ldB16(const _Float16* p) {
    v16h r;
#pragma unroll
    for (int i = 0; i < 8; ++i) { r[i] = p[i * 32]; r[i + 8] = p[(16 + i) * 32]; }
    return r;
}

__device__ __forceinline__ v8f wmma_f16(v16h a, v16h b, v8f c) {
    return __builtin_amdgcn_wmma_f32_16x16x32_f16(false, a, false, b, (short)0, c,
                                                  false, false);
}

// ---------------------------------------------------------------------------
// Tensor Data Mover: issue a 2D f16 tile DMA Global -> LDS (no wait).
// D# per CDNA5 ISA ch.8: group0 = {count/type/addr}, group1 = {dims/strides}.
// ---------------------------------------------------------------------------
#if defined(__has_builtin)
#if __has_builtin(__builtin_amdgcn_tensor_load_to_lds)
#define HAVE_TDM 1
#endif
#endif

__device__ __forceinline__ void tdm_issue_tile_f16(unsigned lds_off, const _Float16* gptr,
                                                   unsigned tensor_w, unsigned tensor_h,
                                                   unsigned stride_elems,
                                                   unsigned tile_w, unsigned tile_h) {
#ifdef HAVE_TDM
    unsigned long long ga = (unsigned long long)(uintptr_t)gptr;
    v4u g0;
    g0[0] = 1u;                                            // count=1, user descriptor
    g0[1] = lds_off;                                       // lds_addr (bytes)
    g0[2] = (unsigned)(ga & 0xffffffffu);                  // global_addr[31:0]
    g0[3] = (unsigned)((ga >> 32) & 0x01ffffffu) | (2u << 30); // addr[56:32] | type=2
    v8i g1;
    g1[0] = (int)(1u << 16);                               // wg_mask=0, data_size=1 (2B)
    g1[1] = (int)((tensor_w & 0xffffu) << 16);             // tensor_dim0[15:0]
    g1[2] = (int)(((tensor_w >> 16) & 0xffffu) | ((tensor_h & 0xffffu) << 16));
    g1[3] = (int)(((tensor_h >> 16) & 0xffffu) | ((tile_w & 0xffffu) << 16));
    g1[4] = (int)(tile_h & 0xffffu);                       // tile_dim1; tile_dim2 = 0
    g1[5] = (int)stride_elems;                             // tensor_dim0_stride[31:0]
    g1[6] = 0;
    g1[7] = 0;
    v4i z = {0, 0, 0, 0};
#if __clang_major__ >= 23
    v8i z8 = {0, 0, 0, 0, 0, 0, 0, 0};
    __builtin_amdgcn_tensor_load_to_lds(g0, g1, z, z, z8, 0);
#else
    __builtin_amdgcn_tensor_load_to_lds(g0, g1, z, z, 0);
#endif
#else
    (void)lds_off; (void)gptr; (void)tensor_w; (void)tensor_h;
    (void)stride_elems; (void)tile_w; (void)tile_h;
#endif
}

__device__ __forceinline__ void tdm_wait_all() {
#ifdef HAVE_TDM
    asm volatile("s_wait_tensorcnt 0x0" ::: "memory");
#endif
}

// Cooperative fallback: 32x32 f16 tile copy (256 threads, 4 f16 each).
__device__ __forceinline__ void coop_copy_tile(_Float16* dst, const _Float16* g,
                                               unsigned stride_elems, int tid) {
    int o = tid * 4;                                       // 0..1023
    int r = o >> 5, c = o & 31;
    *(v4h*)(dst + o) = *(const v4h*)(g + (size_t)r * stride_elems + c);
}

// ---------------------------------------------------------------------------
// 1) styles = w @ (aff_w^T / sqrt(512)) + aff_b         (8 x 1024, tiny)
// ---------------------------------------------------------------------------
__global__ __launch_bounds__(256) void styles_kernel(const float* __restrict__ w,
                                                     const float* __restrict__ aff_w,
                                                     const float* __restrict__ aff_b,
                                                     float* __restrict__ styles) {
    int t = blockIdx.x * 256 + threadIdx.x;          // 0..8191
    int b = t >> 10, n = t & 1023;
    const float* wr = w + b * HDIM;
    const float* ar = aff_w + (size_t)n * HDIM;
    float acc = 0.f;
#pragma unroll 4
    for (int i = 0; i < HDIM; ++i) acc += wr[i] * ar[i];
    styles[t] = acc * 0.044194173824159216f + aff_b[n];   // 1/sqrt(512)
}

// ---------------------------------------------------------------------------
// 2) transpose-convert the five HxH weights to f16 (B-fragment friendly)
// ---------------------------------------------------------------------------
__global__ __launch_bounds__(256) void wcvt_kernel(const float* __restrict__ qw,
                                                   const float* __restrict__ kw,
                                                   const float* __restrict__ vw,
                                                   const float* __restrict__ ww,
                                                   const float* __restrict__ uw,
                                                   _Float16* __restrict__ wt) {
    int which = blockIdx.y;
    const float* src = which == 0 ? qw : which == 1 ? kw : which == 2 ? vw
                       : which == 3 ? ww : uw;
    _Float16* dst = wt + (size_t)which * HDIM * HDIM;
    int t = blockIdx.x * 256 + threadIdx.x;          // 0..262143
    int n = t >> 9, k = t & 511;
    dst[(size_t)k * HDIM + n] = (_Float16)src[(size_t)n * HDIM + k];
}

// ---------------------------------------------------------------------------
// 3) demod factors: rsqrt(sum_i W[n,i]^2 * s[b,i]^2 + eps)  -> [5][B][512]
// ---------------------------------------------------------------------------
__global__ __launch_bounds__(256) void demod_kernel(const float* __restrict__ qw,
                                                    const float* __restrict__ kw,
                                                    const float* __restrict__ vw,
                                                    const float* __restrict__ ww,
                                                    const float* __restrict__ uw,
                                                    const float* __restrict__ styles,
                                                    float* __restrict__ demod) {
    int t = blockIdx.x * 256 + threadIdx.x;          // 0..20479
    int which = t / (BDIM * HDIM);
    int rem   = t % (BDIM * HDIM);
    int b = rem >> 9, n = rem & 511;
    const float* W = which == 0 ? qw : which == 1 ? kw : which == 2 ? vw
                     : which == 3 ? ww : uw;
    const float* st = styles + b * (2 * HDIM) + (which >= 3 ? HDIM : 0);
    const float* wr = W + (size_t)n * HDIM;
    float acc = 0.f;
#pragma unroll 4
    for (int i = 0; i < HDIM; ++i) {
        float wv = wr[i], sv = st[i];
        acc += wv * wv * sv * sv;
    }
    demod[t] = rsqrtf(acc + DEMOD_EPS);
}

// ---------------------------------------------------------------------------
// 4) x * s1, then instance-norm over H; output f16 activations
// ---------------------------------------------------------------------------
__global__ __launch_bounds__(256) void modnorm_kernel(const float* __restrict__ x,
                                                      const float* __restrict__ styles,
                                                      _Float16* __restrict__ xn) {
    __shared__ float rsum[256];
    __shared__ float rsq[256];
    int row = blockIdx.x;                            // 0..8191
    int b = row >> 10;
    int tid = threadIdx.x;
    const float* xr = x + (size_t)row * HDIM;
    const float* s1 = styles + b * (2 * HDIM);
    float v0 = xr[tid] * s1[tid];
    float v1 = xr[tid + 256] * s1[tid + 256];
    rsum[tid] = v0 + v1;
    rsq[tid]  = v0 * v0 + v1 * v1;
    __syncthreads();
#pragma unroll
    for (int s = 128; s > 0; s >>= 1) {
        if (tid < s) { rsum[tid] += rsum[tid + s]; rsq[tid] += rsq[tid + s]; }
        __syncthreads();
    }
    float mu = rsum[0] * (1.f / HDIM);
    float var = rsq[0] * (1.f / HDIM) - mu * mu;
    float rs = rsqrtf(var + IN_EPS);
    _Float16* o = xn + (size_t)row * HDIM;
    o[tid]       = (_Float16)((v0 - mu) * rs);
    o[tid + 256] = (_Float16)((v1 - mu) * rs);
}

// ---------------------------------------------------------------------------
// 5) QKV WMMA GEMM: 2x2 register blocking; shared 32Kx32N weight tile staged
//    by the Tensor Data Mover into ping-pong LDS buffers (DMA for tile k+1
//    overlaps the 4 WMMAs on tile k).
//    grid = (16 Ntiles(32), 32 Mgroups(256), 3 which), block = 256 (8 waves).
// ---------------------------------------------------------------------------
__global__ __launch_bounds__(256) void qkv_wmma_kernel(const _Float16* __restrict__ xn,
                                                       const _Float16* __restrict__ wt,
                                                       const float* __restrict__ demod,
                                                       const float* __restrict__ styles,
                                                       _Float16* __restrict__ qmat,
                                                       _Float16* __restrict__ kT,
                                                       _Float16* __restrict__ vmat,
                                                       _Float16* __restrict__ vflat) {
    __shared__ __align__(16) _Float16 bsh[2][32 * 32];   // ping-pong [k][n] tiles
    int tid = threadIdx.x;
    int lane = tid & 31, wave = tid >> 5;
    int half = lane >> 4, l16 = lane & 15;
    int which = blockIdx.z;
    int n0 = blockIdx.x * 32;
    int m0 = blockIdx.y * 256 + wave * 32;
    const _Float16* Wb = wt + (size_t)which * HDIM * HDIM;
    const _Float16* arow0 = xn + (size_t)(m0 + l16) * HDIM;
    const _Float16* arow1 = xn + (size_t)(m0 + 16 + l16) * HDIM;
    v8f acc00 = {}, acc01 = {}, acc10 = {}, acc11 = {};

    const int NT = HDIM / 32;
#ifdef HAVE_TDM
    if (wave == 0)
        tdm_issue_tile_f16((unsigned)(uintptr_t)&bsh[0][0], Wb + n0,
                           HDIM, HDIM, HDIM, 32, 32);
#else
    coop_copy_tile(&bsh[0][0], Wb + n0, HDIM, tid);
#endif
    for (int kk = 0; kk < NT; ++kk) {
        int cur = kk & 1;
#ifdef HAVE_TDM
        if (wave == 0) tdm_wait_all();               // tile kk landed in LDS
#endif
        __syncthreads();
        if (kk + 1 < NT) {                           // prefetch tile kk+1
            const _Float16* gnext = Wb + (size_t)((kk + 1) * 32) * HDIM + n0;
#ifdef HAVE_TDM
            if (wave == 0)
                tdm_issue_tile_f16((unsigned)(uintptr_t)&bsh[cur ^ 1][0], gnext,
                                   HDIM, HDIM, HDIM, 32, 32);
#else
            coop_copy_tile(&bsh[cur ^ 1][0], gnext, HDIM, tid);
#endif
            __builtin_prefetch(arow0 + (kk + 1) * 32, 0, 1);
            __builtin_prefetch(arow1 + (kk + 1) * 32, 0, 1);
        }
        v16h a0 = ldA(arow0 + kk * 32, half);
        v16h a1 = ldA(arow1 + kk * 32, half);
        const _Float16* brow = &bsh[cur][half * 8 * 32 + l16];
        v16h b0 = ldB16(brow);
        v16h b1 = ldB16(brow + 16);
        acc00 = wmma_f16(a0, b0, acc00);
        acc01 = wmma_f16(a0, b1, acc01);
        acc10 = wmma_f16(a1, b0, acc10);
        acc11 = wmma_f16(a1, b1, acc11);
    }

    int b = m0 >> 10;
    const float* dmd = demod + which * (BDIM * HDIM) + b * HDIM;
#pragma unroll
    for (int j = 0; j < 2; ++j) {
        int n = n0 + j * 16 + l16;
        float scale = dmd[n];
        if (which == 2) scale *= styles[b * (2 * HDIM) + HDIM + n];
        int h = n >> 5, d = n & 31;
#pragma unroll
        for (int i = 0; i < 2; ++i) {
            v8f acc = (i == 0) ? (j == 0 ? acc00 : acc01)
                               : (j == 0 ? acc10 : acc11);
#pragma unroll
            for (int r = 0; r < 8; ++r) {
                int m = m0 + i * 16 + r + 8 * half;
                int s = m & (SQ - 1);
                float val = acc[r] * scale;
                if (which == 0) {
                    qmat[(((size_t)(b * HEADS + h) * SQ) + s) * DEPTH + d] = (_Float16)val;
                } else if (which == 1) {
                    kT[(((size_t)(b * HEADS + h) * DEPTH) + d) * SQ + s] = (_Float16)val;
                } else {
                    vmat[(((size_t)(b * HEADS + h) * SQ) + s) * DEPTH + d] = (_Float16)val;
                    vflat[(size_t)m * HDIM + n] = (_Float16)val;
                }
            }
        }
    }
}

// ---------------------------------------------------------------------------
// 6) Flash attention, 32-key tiles. All 8 waves of a block share the same
//    K-tile (32 depth x 32 keys of kT) and V-tile (32 keys x 32 depth), so
//    both are TDM-staged into ping-pong LDS buffers (8x less global traffic),
//    DMA overlapped with the 4 WMMAs + softmax of the current tile.
// ---------------------------------------------------------------------------
__global__ __launch_bounds__(256) void attn_kernel(const _Float16* __restrict__ qmat,
                                                   const _Float16* __restrict__ kT,
                                                   const _Float16* __restrict__ vmat,
                                                   _Float16* __restrict__ attn_out) {
    __shared__ float pbuf[8][16 * 32];
    __shared__ __align__(16) _Float16 ksh[2][32 * 32];   // [depth][key]
    __shared__ __align__(16) _Float16 vsh[2][32 * 32];   // [key][depth]
    int tid = threadIdx.x;
    int lane = tid & 31, wave = tid >> 5;
    int half = lane >> 4, l16 = lane & 15;
    int bh = blockIdx.x;                             // 0..127
    int b = bh >> 4, h = bh & 15;
    int q0 = blockIdx.y * 128 + wave * 16;
    const _Float16* qb = qmat + (size_t)bh * SQ * DEPTH;
    const _Float16* kb = kT + (size_t)bh * DEPTH * SQ;
    const _Float16* vb = vmat + (size_t)bh * SQ * DEPTH;

    v16h qfrag = ldA(qb + (size_t)(q0 + l16) * DEPTH, half);
    v8f o0 = {}, o1 = {};
    float mrun[8], lrun[8];
#pragma unroll
    for (int r = 0; r < 8; ++r) { mrun[r] = -1e30f; lrun[r] = 0.f; }
    const float scl = 0.17677669529663687f;          // 1/sqrt(32)

    const int NT = SQ / 32;
#ifdef HAVE_TDM
    if (wave == 0) {
        tdm_issue_tile_f16((unsigned)(uintptr_t)&ksh[0][0], kb, SQ, DEPTH, SQ, 32, 32);
        tdm_issue_tile_f16((unsigned)(uintptr_t)&vsh[0][0], vb, DEPTH, SQ, DEPTH, 32, 32);
    }
#else
    coop_copy_tile(&ksh[0][0], kb, SQ, tid);
    coop_copy_tile(&vsh[0][0], vb, DEPTH, tid);
#endif
    for (int kt = 0; kt < NT; ++kt) {
        int cur = kt & 1;
#ifdef HAVE_TDM
        if (wave == 0) tdm_wait_all();               // K/V tiles kt landed
#endif
        __syncthreads();
        if (kt + 1 < NT) {                           // prefetch tile kt+1
            const _Float16* kg = kb + (kt + 1) * 32;
            const _Float16* vg = vb + (size_t)((kt + 1) * 32) * DEPTH;
#ifdef HAVE_TDM
            if (wave == 0) {
                tdm_issue_tile_f16((unsigned)(uintptr_t)&ksh[cur ^ 1][0], kg,
                                   SQ, DEPTH, SQ, 32, 32);
                tdm_issue_tile_f16((unsigned)(uintptr_t)&vsh[cur ^ 1][0], vg,
                                   DEPTH, SQ, DEPTH, 32, 32);
            }
#else
            coop_copy_tile(&ksh[cur ^ 1][0], kg, SQ, tid);
            coop_copy_tile(&vsh[cur ^ 1][0], vg, DEPTH, tid);
#endif
        }

        // Kᵀ B-fragments from LDS: lane n = l16 (key col), elements over depth k
        const _Float16* krow = &ksh[cur][half * 8 * 32 + l16];
        v16h kf0 = ldB16(krow);
        v16h kf1 = ldB16(krow + 16);
        v8f s0 = {}, s1 = {};
        s0 = wmma_f16(qfrag, kf0, s0);
        s1 = wmma_f16(qfrag, kf1, s1);

        float corr[8];
#pragma unroll
        for (int r = 0; r < 8; ++r) {
            float va = s0[r] * scl;
            float vc = s1[r] * scl;
            float mx = fmaxf(va, vc);
            mx = fmaxf(mx, __shfl_xor(mx, 1));
            mx = fmaxf(mx, __shfl_xor(mx, 2));
            mx = fmaxf(mx, __shfl_xor(mx, 4));
            mx = fmaxf(mx, __shfl_xor(mx, 8));
            float mnew = fmaxf(mrun[r], mx);
            float p0 = __expf(va - mnew);
            float p1 = __expf(vc - mnew);
            float ps = p0 + p1;
            ps += __shfl_xor(ps, 1);
            ps += __shfl_xor(ps, 2);
            ps += __shfl_xor(ps, 4);
            ps += __shfl_xor(ps, 8);
            float c = __expf(mrun[r] - mnew);
            lrun[r] = lrun[r] * c + ps;
            mrun[r] = mnew;
            corr[r] = c;
            pbuf[wave][(r + 8 * half) * 32 + l16] = p0;        // C layout -> LDS
            pbuf[wave][(r + 8 * half) * 32 + 16 + l16] = p1;
        }
        asm volatile("s_wait_dscnt 0" ::: "memory");           // wave-local LDS sync

#pragma unroll
        for (int r = 0; r < 8; ++r) { o0[r] *= corr[r]; o1[r] *= corr[r]; }

        // P as A-fragment: row = l16; e<8 -> k = half*8+e, e>=8 -> k = 16+half*8+(e-8)
        v16h pfrag;
#pragma unroll
        for (int e = 0; e < 8; ++e) {
            pfrag[e]     = (_Float16)pbuf[wave][l16 * 32 + half * 8 + e];
            pfrag[e + 8] = (_Float16)pbuf[wave][l16 * 32 + 16 + half * 8 + e];
        }
        // V B-fragments from LDS: lane n = l16 (depth col), elements over key k
        const _Float16* vrow = &vsh[cur][half * 8 * 32 + l16];
        v16h v0f = ldB16(vrow);
        v16h v1f = ldB16(vrow + 16);
        o0 = wmma_f16(pfrag, v0f, o0);
        o1 = wmma_f16(pfrag, v1f, o1);
    }

#pragma unroll
    for (int r = 0; r < 8; ++r) {
        int s = q0 + r + 8 * half;
        float inv = 1.f / lrun[r];
        size_t base = ((size_t)(b * SQ + s)) * HDIM + h * DEPTH;
        attn_out[base + l16]      = (_Float16)(o0[r] * inv);
        attn_out[base + 16 + l16] = (_Float16)(o1[r] * inv);
    }
}

// ---------------------------------------------------------------------------
// 7) Fused out-proj + skip-proj + noise + bias + lrelu + clamp, 2x2 blocking,
//    both weight tiles TDM-staged into ping-pong LDS buffers.
// ---------------------------------------------------------------------------
__global__ __launch_bounds__(256) void outskip_wmma_kernel(const _Float16* __restrict__ attn_out,
                                                           const _Float16* __restrict__ vflat,
                                                           const _Float16* __restrict__ wt,
                                                           const float* __restrict__ demod,
                                                           const float* __restrict__ noise_const,
                                                           const float* __restrict__ noise_strength,
                                                           const float* __restrict__ bias,
                                                           float* __restrict__ out) {
    __shared__ __align__(16) _Float16 wsh[2][32 * 32];   // ping-pong w-proj tiles
    __shared__ __align__(16) _Float16 ush[2][32 * 32];   // ping-pong u-proj tiles
    int tid = threadIdx.x;
    int lane = tid & 31, wave = tid >> 5;
    int half = lane >> 4, l16 = lane & 15;
    int n0 = blockIdx.x * 32;
    int m0 = blockIdx.y * 256 + wave * 32;
    const _Float16* Ww = wt + (size_t)3 * HDIM * HDIM;
    const _Float16* Wu = wt + (size_t)4 * HDIM * HDIM;
    const _Float16* a1r0 = attn_out + (size_t)(m0 + l16) * HDIM;
    const _Float16* a1r1 = attn_out + (size_t)(m0 + 16 + l16) * HDIM;
    const _Float16* a2r0 = vflat + (size_t)(m0 + l16) * HDIM;
    const _Float16* a2r1 = vflat + (size_t)(m0 + 16 + l16) * HDIM;
    v8f w00 = {}, w01 = {}, w10 = {}, w11 = {};
    v8f u00 = {}, u01 = {}, u10 = {}, u11 = {};

    const int NT = HDIM / 32;
#ifdef HAVE_TDM
    if (wave == 0) {
        tdm_issue_tile_f16((unsigned)(uintptr_t)&wsh[0][0], Ww + n0, HDIM, HDIM, HDIM, 32, 32);
        tdm_issue_tile_f16((unsigned)(uintptr_t)&ush[0][0], Wu + n0, HDIM, HDIM, HDIM, 32, 32);
    }
#else
    coop_copy_tile(&wsh[0][0], Ww + n0, HDIM, tid);
    coop_copy_tile(&ush[0][0], Wu + n0, HDIM, tid);
#endif
    for (int kk = 0; kk < NT; ++kk) {
        int cur = kk & 1;
#ifdef HAVE_TDM
        if (wave == 0) tdm_wait_all();
#endif
        __syncthreads();
        if (kk + 1 < NT) {
            size_t goff = (size_t)((kk + 1) * 32) * HDIM + n0;
#ifdef HAVE_TDM
            if (wave == 0) {
                tdm_issue_tile_f16((unsigned)(uintptr_t)&wsh[cur ^ 1][0], Ww + goff,
                                   HDIM, HDIM, HDIM, 32, 32);
                tdm_issue_tile_f16((unsigned)(uintptr_t)&ush[cur ^ 1][0], Wu + goff,
                                   HDIM, HDIM, HDIM, 32, 32);
            }
#else
            coop_copy_tile(&wsh[cur ^ 1][0], Ww + goff, HDIM, tid);
            coop_copy_tile(&ush[cur ^ 1][0], Wu + goff, HDIM, tid);
#endif
            __builtin_prefetch(a1r0 + (kk + 1) * 32, 0, 1);
            __builtin_prefetch(a2r0 + (kk + 1) * 32, 0, 1);
        }
        const _Float16* wrow = &wsh[cur][half * 8 * 32 + l16];
        const _Float16* urow = &ush[cur][half * 8 * 32 + l16];
        v16h bw0 = ldB16(wrow);
        v16h bw1 = ldB16(wrow + 16);
        v16h bu0 = ldB16(urow);
        v16h bu1 = ldB16(urow + 16);
        v16h a10 = ldA(a1r0 + kk * 32, half);
        v16h a11 = ldA(a1r1 + kk * 32, half);
        v16h a20 = ldA(a2r0 + kk * 32, half);
        v16h a21 = ldA(a2r1 + kk * 32, half);
        w00 = wmma_f16(a10, bw0, w00);
        w01 = wmma_f16(a10, bw1, w01);
        w10 = wmma_f16(a11, bw0, w10);
        w11 = wmma_f16(a11, bw1, w11);
        u00 = wmma_f16(a20, bu0, u00);
        u01 = wmma_f16(a20, bu1, u01);
        u10 = wmma_f16(a21, bu0, u10);
        u11 = wmma_f16(a21, bu1, u11);
    }
    int b = m0 >> 10;
    float ns = noise_strength[0];
#pragma unroll
    for (int j = 0; j < 2; ++j) {
        int n = n0 + j * 16 + l16;
        float wd = demod[3 * (BDIM * HDIM) + b * HDIM + n];
        float ud = demod[4 * (BDIM * HDIM) + b * HDIM + n];
        float bn = bias[n];
#pragma unroll
        for (int i = 0; i < 2; ++i) {
            v8f aw = (i == 0) ? (j == 0 ? w00 : w01) : (j == 0 ? w10 : w11);
            v8f au = (i == 0) ? (j == 0 ? u00 : u01) : (j == 0 ? u10 : u11);
#pragma unroll
            for (int r = 0; r < 8; ++r) {
                int m = m0 + i * 16 + r + 8 * half;
                int s = m & (SQ - 1);
                float val = aw[r] * wd + au[r] * ud + noise_const[s] * ns + bn;
                val = (val >= 0.f) ? val : LRELU * val;
                val = fminf(fmaxf(val, -CLAMPV), CLAMPV);
                out[(size_t)m * HDIM + n] = val;
            }
        }
    }
}

// ---------------------------------------------------------------------------
// launcher
// ---------------------------------------------------------------------------
extern "C" void kernel_launch(void* const* d_in, const int* in_sizes, int n_in,
                              void* d_out, int out_size, void* d_ws, size_t ws_size,
                              hipStream_t stream) {
    const float* x        = (const float*)d_in[0];
    const float* w        = (const float*)d_in[1];
    const float* qw       = (const float*)d_in[2];
    const float* kw       = (const float*)d_in[3];
    const float* vw       = (const float*)d_in[4];
    const float* ww       = (const float*)d_in[5];
    const float* uw       = (const float*)d_in[6];
    const float* aff_w    = (const float*)d_in[7];
    const float* aff_b    = (const float*)d_in[8];
    const float* noise_c  = (const float*)d_in[9];
    const float* noise_s  = (const float*)d_in[10];
    const float* bias     = (const float*)d_in[11];
    float* out = (float*)d_out;

    char* ws = (char*)d_ws;
    size_t off = 0;
    auto alloc = [&](size_t bytes) { char* p = ws + off; off = (off + bytes + 255) & ~(size_t)255; return p; };
    float*    styles   = (float*)   alloc(BDIM * 2 * HDIM * sizeof(float));
    float*    demod    = (float*)   alloc(5 * BDIM * HDIM * sizeof(float));
    _Float16* wt       = (_Float16*)alloc((size_t)5 * HDIM * HDIM * sizeof(_Float16));
    _Float16* xn       = (_Float16*)alloc((size_t)ROWS * HDIM * sizeof(_Float16));
    _Float16* qmat     = (_Float16*)alloc((size_t)ROWS * HDIM * sizeof(_Float16));
    _Float16* kT       = (_Float16*)alloc((size_t)ROWS * HDIM * sizeof(_Float16));
    _Float16* vmat     = (_Float16*)alloc((size_t)ROWS * HDIM * sizeof(_Float16));
    _Float16* vflat    = (_Float16*)alloc((size_t)ROWS * HDIM * sizeof(_Float16));
    _Float16* attn_out = (_Float16*)alloc((size_t)ROWS * HDIM * sizeof(_Float16));
    (void)ws_size; (void)n_in; (void)in_sizes; (void)out_size;

    styles_kernel<<<32, 256, 0, stream>>>(w, aff_w, aff_b, styles);
    wcvt_kernel<<<dim3(1024, 5), 256, 0, stream>>>(qw, kw, vw, ww, uw, wt);
    demod_kernel<<<80, 256, 0, stream>>>(qw, kw, vw, ww, uw, styles, demod);
    modnorm_kernel<<<ROWS, 256, 0, stream>>>(x, styles, xn);
    qkv_wmma_kernel<<<dim3(HDIM / 32, ROWS / 256, 3), 256, 0, stream>>>(
        xn, wt, demod, styles, qmat, kT, vmat, vflat);
    attn_kernel<<<dim3(BDIM * HEADS, SQ / 128), 256, 0, stream>>>(
        qmat, kT, vmat, attn_out);
    outskip_wmma_kernel<<<dim3(HDIM / 32, ROWS / 256), 256, 0, stream>>>(
        attn_out, vflat, wt, demod, noise_c, noise_s, bias, out);
}
